// FlashAttentionModule_446676599322
// MI455X (gfx1250) — matrix operand order repaired and checked by reading, not compiled, not run on hardware
//
#include <hip/hip_runtime.h>

// Sliding-window GQA flash attention forward, f32 in/out, f16 WMMA math.
// B=2, S=4096, H_Q=32, H_KV=8, D=128, BLK=128, WIN=127 (band = 2 key blocks).

typedef _Float16 v16h __attribute__((ext_vector_type(16)));
typedef _Float16 v8h  __attribute__((ext_vector_type(8)));
typedef _Float16 h4   __attribute__((ext_vector_type(4)));
typedef float    v8f  __attribute__((ext_vector_type(8)));

#define SEQ   4096
#define HQ    32
#define HKV   8
#define DH    128
#define NBLK  32          // SEQ / 128
#define KSTR  136         // K LDS row stride in halves (128 + 8 pad)
#define VSTR  264         // Vt/P LDS row stride in halves (256 + 8 pad)

static __device__ __forceinline__ v16h cat8(v8h a, v8h b) {
  return __builtin_shufflevector(a, b, 0,1,2,3,4,5,6,7,8,9,10,11,12,13,14,15);
}

__global__ __launch_bounds__(256, 1)
void swa_attn_fwd(const float* __restrict__ Q, const float* __restrict__ K,
                  const float* __restrict__ V, float* __restrict__ O) {
  // KPl: K window (256 x 128 halves) during QK; reused for P (128 x 256) during PV.
  __shared__ _Float16 KPl[256 * KSTR];   // 69,632 B
  __shared__ _Float16 Vt [128 * VSTR];   // 67,584 B  (V transposed: row=d, col=key)

  const int tid  = threadIdx.x;
  const int lane = tid & 31;
  const int w    = tid >> 5;     // wave id 0..7 -> 16-row query strip
  const int lo   = lane & 15;
  const int hi   = lane >> 4;

  int idx = blockIdx.x;
  const int nb  = idx & (NBLK - 1); idx >>= 5;
  const int hq  = idx & (HQ - 1);   idx >>= 5;
  const int b   = idx;
  const int hkv = hq >> 2;          // g = 4

  // ---------------- cooperative load: K window -> KPl (f16), V window -> Vt (f16, transposed)
  const long kvBase = ((long)b * SEQ) * (HKV * DH) + (long)hkv * DH;
  #pragma unroll 4
  for (int i = 0; i < 32; ++i) {
    const int i4  = tid + i * 256;      // float4 index over 256*128/4
    const int key = i4 >> 5;            // window-local key row 0..255
    const int d   = (i4 & 31) << 2;     // d offset 0..124 step 4
    const int s   = (nb - 1) * 128 + key;
    float4 kq = {0.f, 0.f, 0.f, 0.f};
    float4 vq = {0.f, 0.f, 0.f, 0.f};
    if (s >= 0) {
      const long off = kvBase + (long)s * (HKV * DH) + d;
      kq = *reinterpret_cast<const float4*>(K + off);
      vq = *reinterpret_cast<const float4*>(V + off);
    }
    h4 kh;
    kh[0] = (_Float16)kq.x; kh[1] = (_Float16)kq.y;
    kh[2] = (_Float16)kq.z; kh[3] = (_Float16)kq.w;
    *reinterpret_cast<h4*>(&KPl[key * KSTR + d]) = kh;
    Vt[(d + 0) * VSTR + key] = (_Float16)vq.x;
    Vt[(d + 1) * VSTR + key] = (_Float16)vq.y;
    Vt[(d + 2) * VSTR + key] = (_Float16)vq.z;
    Vt[(d + 3) * VSTR + key] = (_Float16)vq.w;
  }
  __syncthreads();

  // ---------------- per-lane Q load into WMMA A layout (4 k-steps over D=128)
  const int qrow = nb * 128 + w * 16 + lo;                       // this lane's M row
  const float* qptr = Q + (((long)b * SEQ + qrow) * HQ + hq) * DH;
  v16h qA[4];
  #pragma unroll
  for (int ks = 0; ks < 4; ++ks) {
    const int d0 = ks * 32 + hi * 8;           // lo lanes: K 0..7 / 16..23; hi: 8..15 / 24..31
    const float4* p0 = reinterpret_cast<const float4*>(qptr + d0);
    const float4* p1 = reinterpret_cast<const float4*>(qptr + d0 + 16);
    float f[16];
    float4 a0 = p0[0], a1 = p0[1], b0 = p1[0], b1 = p1[1];
    f[0]=a0.x; f[1]=a0.y; f[2]=a0.z; f[3]=a0.w;
    f[4]=a1.x; f[5]=a1.y; f[6]=a1.z; f[7]=a1.w;
    f[8]=b0.x; f[9]=b0.y; f[10]=b0.z; f[11]=b0.w;
    f[12]=b1.x; f[13]=b1.y; f[14]=b1.z; f[15]=b1.w;
    v16h a;
    #pragma unroll
    for (int e = 0; e < 16; ++e) a[e] = (_Float16)f[e];
    qA[ks] = a;
  }

  // ---------------- scores: S[16x256] per wave = Q strip x K^T  (64 WMMAs)
  // B-tile fetch: lane = key col (16t+lo); lo lanes: d 0..7 / 16..23; hi: 8..15 / 24..31.
  // Software-pipelined with a depth-4 rotating buffer so ds loads overlap WMMAs.
  auto loadKB = [&](int i) -> v16h {
    const int t  = i >> 2;
    const int ks = i & 3;
    const int key = 16 * t + lo;
    const int ds  = ks * 32 + hi * 8;
    const v8h* bp = reinterpret_cast<const v8h*>(&KPl[key * KSTR + ds]);
    return cat8(bp[0], bp[2]);
  };

  v8f acc[16];
  {
    v16h kb[4];
    #pragma unroll
    for (int i = 0; i < 4; ++i) kb[i] = loadKB(i);
    #pragma unroll
    for (int i = 0; i < 64; ++i) {
      const int t  = i >> 2;
      const int ks = i & 3;
      const v16h cur = kb[i & 3];
      if (i + 4 < 64) kb[i & 3] = loadKB(i + 4);
      const v8f cin = (ks == 0) ? (v8f){0.f,0.f,0.f,0.f,0.f,0.f,0.f,0.f} : acc[t];
      acc[t] = __builtin_amdgcn_wmma_f32_16x16x32_f16(false, qA[ks], false, cur,
                                                      (short)0, cin, false, false);
    }
  }
  __syncthreads();   // all waves done reading KPl before it is reused for P

  // ---------------- band mask + softmax (rows striped: m = r + 8*hi, col = 16*t + lo)
  const float scale = 0.08838834764831845f;    // 1/sqrt(128)
  #pragma unroll
  for (int r = 0; r < 8; ++r) {
    const int m  = r + 8 * hi;
    const int qr = w * 16 + m;                 // block-local query row 0..127
    float mx = -1e30f;
    #pragma unroll
    for (int t = 0; t < 16; ++t) {
      const int c = 16 * t + lo;
      float sv = acc[t][r] * scale;
      const bool valid = (c >= qr + 1) && (c <= qr + 128) && (nb > 0 || c >= 128);
      sv = valid ? sv : -1e30f;
      acc[t][r] = sv;
      mx = fmaxf(mx, sv);
    }
    mx = fmaxf(mx, __shfl_xor(mx, 1));
    mx = fmaxf(mx, __shfl_xor(mx, 2));
    mx = fmaxf(mx, __shfl_xor(mx, 4));
    mx = fmaxf(mx, __shfl_xor(mx, 8));
    float sm = 0.f;
    #pragma unroll
    for (int t = 0; t < 16; ++t) {
      const float e = __expf(acc[t][r] - mx);
      acc[t][r] = e;
      sm += e;
    }
    sm += __shfl_xor(sm, 1);
    sm += __shfl_xor(sm, 2);
    sm += __shfl_xor(sm, 4);
    sm += __shfl_xor(sm, 8);
    const float inv = 1.f / sm;
    #pragma unroll
    for (int t = 0; t < 16; ++t) {            // normalized P -> LDS (per-wave strip of KPl)
      KPl[(w * 16 + m) * VSTR + 16 * t + lo] = (_Float16)(acc[t][r] * inv);
    }
  }

  // ---------------- out = P x Vwin  (64 WMMAs), K-dim = 256 keys in 8 steps
  auto loadPA = [&](int ks) -> v16h {
    const int k0 = ks * 32 + hi * 8;
    const v8h* pa = reinterpret_cast<const v8h*>(&KPl[(w * 16 + lo) * VSTR + k0]);
    const v8h* pb = reinterpret_cast<const v8h*>(&KPl[(w * 16 + lo) * VSTR + k0 + 16]);
    return cat8(pa[0], pb[0]);
  };
  auto loadVB = [&](int i) -> v16h {
    const int ks = i >> 3;
    const int nt = i & 7;
    const int dcol = 16 * nt + lo;
    const int kst  = ks * 32 + hi * 8;
    const v8h* vp = reinterpret_cast<const v8h*>(&Vt[dcol * VSTR + kst]);
    return cat8(vp[0], vp[2]);
  };

  v8f o[8];
  #pragma unroll
  for (int nt = 0; nt < 8; ++nt) o[nt] = (v8f){0.f,0.f,0.f,0.f,0.f,0.f,0.f,0.f};
  {
    v16h vb[4];
    #pragma unroll
    for (int i = 0; i < 4; ++i) vb[i] = loadVB(i);
    v16h aCur = loadPA(0);
    v16h aNext = aCur;
    #pragma unroll
    for (int i = 0; i < 64; ++i) {
      const int ks = i >> 3;
      const int nt = i & 7;
      if (nt == 0 && ks > 0) aCur = aNext;
      if (nt == 0 && ks + 1 < 8) aNext = loadPA(ks + 1);   // prefetch next A strip
      const v16h cur = vb[i & 3];
      if (i + 4 < 64) vb[i & 3] = loadVB(i + 4);
      o[nt] = __builtin_amdgcn_wmma_f32_16x16x32_f16(false, aCur, false, cur,
                                                     (short)0, o[nt], false, false);
    }
  }

  // ---------------- store O (f32): row m = r + 8*hi, col d = 16*nt + lo
  #pragma unroll
  for (int r = 0; r < 8; ++r) {
    const int grow = nb * 128 + w * 16 + r + 8 * hi;
    float* op = O + (((long)b * SEQ + grow) * HQ + hq) * DH;
    #pragma unroll
    for (int nt = 0; nt < 8; ++nt) {
      op[16 * nt + lo] = o[nt][r];
    }
  }
}

extern "C" void kernel_launch(void* const* d_in, const int* in_sizes, int n_in,
                              void* d_out, int out_size, void* d_ws, size_t ws_size,
                              hipStream_t stream) {
  (void)in_sizes; (void)n_in; (void)out_size; (void)d_ws; (void)ws_size;
  const float* Q = (const float*)d_in[0];
  const float* K = (const float*)d_in[1];
  const float* V = (const float*)d_in[2];
  float* O = (float*)d_out;
  const int nblocks = 2 * HQ * NBLK;   // B * H_Q * (S/128) = 2048 workgroups
  swa_attn_fwd<<<nblocks, 256, 0, stream>>>(Q, K, V, O);
}
